// LSTMLayer_11647951307304
// MI455X (gfx1250) — matrix operand order repaired and checked by reading, not compiled, not run on hardware
//
#include <hip/hip_runtime.h>

// ---------------------------------------------------------------------------
// LayerNorm LSTM (B=64, T=256, H=1024, L=2) on MI455X (gfx1250, wave32).
//  * one-time: W -> bf16 [N][K] transposed (32MB, L2-resident), X -> bf16
//  * per step/layer: [64,2048]x[2048,4096] recurrent GEMM via
//    v_wmma_f32_16x16x32_bf16, K split 2-ways (x-part / h-part) into two
//    z buffers (no atomics), 32x32 output tile per wave (4 accumulators)
//  * LDS tiles staged with global_load_async_to_lds_b128 (ASYNCcnt-tracked)
//  * fused pointwise: 5x layernorm (shfl+LDS reductions), gates, state
//    update, masking, running time-mean into d_out; emits h in fp32+bf16
// ---------------------------------------------------------------------------

#define B_  64
#define T_  256
#define H_  1024
#define L_  2
#define K_  (2 * H_)      // 2048
#define N_  (4 * H_)      // 4096

#if __has_builtin(__builtin_amdgcn_global_load_async_to_lds_b128) && \
    __has_builtin(__builtin_amdgcn_s_wait_asynccnt)
#define USE_ASYNC_LDS 1
#else
#define USE_ASYNC_LDS 0
#endif

typedef __attribute__((ext_vector_type(16))) __bf16 v16bf;
typedef __attribute__((ext_vector_type(8)))  float  v8f;

// matches the builtin's parameter type: int __vector(4)
typedef __attribute__((__vector_size__(4 * sizeof(int)))) int i4_t;
typedef __attribute__((address_space(1))) i4_t* gvec_p;   // global
typedef __attribute__((address_space(3))) i4_t* lvec_p;   // LDS

union Frag16 {
    v16bf v;
    uint4 q[2];
};

__device__ __forceinline__ unsigned short f2bf(float x) {
    unsigned int u = __float_as_uint(x);
    u += 0x7FFFu + ((u >> 16) & 1u);   // round-to-nearest-even
    return (unsigned short)(u >> 16);
}

__device__ __forceinline__ float sigmoidf_(float x) {
    return 1.0f / (1.0f + __expf(-x));
}

// ---------------------------------------------------------------------------
// One-time: inputs [B][T][H] fp32 -> bf16. grid 16384, block 256.
// ---------------------------------------------------------------------------
__global__ void convert_x(const float* __restrict__ x,
                          unsigned short* __restrict__ xbf) {
    int idx = (blockIdx.x * 256 + threadIdx.x) * 4;
    float4 v = *(const float4*)(x + idx);
    ushort4 o;
    o.x = f2bf(v.x); o.y = f2bf(v.y); o.z = f2bf(v.z); o.w = f2bf(v.w);
    *(ushort4*)(xbf + idx) = o;
}

// ---------------------------------------------------------------------------
// One-time: W [L][K][N] fp32 -> WbT [L][N][K] bf16 (tiled LDS transpose)
// grid (N/32, K/32, L), block (32, 8)
// ---------------------------------------------------------------------------
__global__ void prep_weights(const float* __restrict__ W,
                             unsigned short* __restrict__ WbT) {
    __shared__ float tile[32][33];
    const int l  = blockIdx.z;
    const int n0 = blockIdx.x * 32;
    const int k0 = blockIdx.y * 32;
    const float* Wl = W + (size_t)l * K_ * N_;
#pragma unroll
    for (int j = 0; j < 4; ++j) {
        int k = k0 + threadIdx.y + j * 8;
        tile[threadIdx.y + j * 8][threadIdx.x] =
            Wl[(size_t)k * N_ + n0 + threadIdx.x];
    }
    __syncthreads();
    unsigned short* Ol = WbT + (size_t)l * N_ * K_;
#pragma unroll
    for (int j = 0; j < 4; ++j) {
        int n = n0 + threadIdx.y + j * 8;
        Ol[(size_t)n * K_ + k0 + threadIdx.x] =
            f2bf(tile[threadIdx.x][threadIdx.y + j * 8]);
    }
}

// ---------------------------------------------------------------------------
// Zero recurrent state (fp32 + bf16) and output accumulator. grid 512.
// ---------------------------------------------------------------------------
__global__ void init_state(float* __restrict__ c, float* __restrict__ h,
                           unsigned short* __restrict__ hbf,
                           float* __restrict__ out) {
    int idx = blockIdx.x * 256 + threadIdx.x;
    if (idx < L_ * B_ * H_) { c[idx] = 0.0f; h[idx] = 0.0f; hbf[idx] = 0; }
    if (idx < B_ * H_)      { out[idx] = 0.0f; }
}

// ---------------------------------------------------------------------------
// Recurrent GEMM, K split across gridDim.y (y=0: x-part, y=1: h-part).
// grid (64, 2), block 128 (4 waves). Wave tile 32x32 -> 4 accumulators.
// All operands already bf16; LDS staged via async-to-LDS when available.
// ---------------------------------------------------------------------------
__global__ void __launch_bounds__(128) lstm_gemm(
        const unsigned short* __restrict__ A0, long strideA0,  // K 0..1023
        const unsigned short* __restrict__ A1,                 // K 1024..2047
        const unsigned short* __restrict__ WbTl,               // [N][K] bf16
        float* __restrict__ z) {                               // [2][B][N]
    __shared__ unsigned short Atile[64][72];  // 144B row stride (16B aligned)
    __shared__ unsigned short Btile[64][72];

    const int tid  = threadIdx.x;
    const int lane = tid & 31;
    const int wave = tid >> 5;
    const int wm   = wave & 1;        // M subtile: wm*32
    const int wn   = wave >> 1;       // N subtile: wn*32
    const int half = lane >> 4;
    const int mrow = lane & 15;

    const int ntile0 = blockIdx.x * 64;
    const unsigned short* A = blockIdx.y ? A1 : A0;
    const long sA           = blockIdx.y ? (long)H_ : strideA0;
    const unsigned short* Wk = WbTl + (size_t)blockIdx.y * H_;  // K offset
    float* zout = z + (size_t)blockIdx.y * B_ * N_;

    // staging pointers (constant across K chunks, advanced by kc)
    const unsigned short* gA[4];
    const unsigned short* gB[4];
    unsigned short* lA[4];
    unsigned short* lB[4];
#pragma unroll
    for (int i = 0; i < 4; ++i) {
        int linear = tid + i * 128;      // 0..511
        int row = linear >> 3;           // 0..63
        int k8  = (linear & 7) * 8;      // 0..56
        gA[i] = A + (size_t)row * sA + k8;
        gB[i] = Wk + (size_t)(ntile0 + row) * K_ + k8;
        lA[i] = &Atile[row][k8];
        lB[i] = &Btile[row][k8];
    }

    v8f acc00 = {}, acc01 = {}, acc10 = {}, acc11 = {};

    for (int kc = 0; kc < H_; kc += 64) {
#pragma unroll
        for (int i = 0; i < 4; ++i) {
#if USE_ASYNC_LDS
            __builtin_amdgcn_global_load_async_to_lds_b128(
                (gvec_p)(gA[i] + kc), (lvec_p)lA[i], 0, 0);
            __builtin_amdgcn_global_load_async_to_lds_b128(
                (gvec_p)(gB[i] + kc), (lvec_p)lB[i], 0, 0);
#else
            *(uint4*)lA[i] = *(const uint4*)(gA[i] + kc);
            *(uint4*)lB[i] = *(const uint4*)(gB[i] + kc);
#endif
        }
        if (kc + 64 < H_ && tid < 64) {   // L2 prefetch of next B chunk
            __builtin_prefetch(Wk + (size_t)(ntile0 + tid) * K_ + kc + 64, 0, 1);
        }
#if USE_ASYNC_LDS
        __builtin_amdgcn_s_wait_asynccnt(0);
#endif
        __syncthreads();

#pragma unroll
        for (int kk = 0; kk < 64; kk += 32) {
            Frag16 a0, a1, b0, b1;
            const int ra0 = wm * 32 + mrow;
            const int ra1 = ra0 + 16;
            a0.q[0] = *(const uint4*)&Atile[ra0][kk + half * 8];
            a0.q[1] = *(const uint4*)&Atile[ra0][kk + 16 + half * 8];
            a1.q[0] = *(const uint4*)&Atile[ra1][kk + half * 8];
            a1.q[1] = *(const uint4*)&Atile[ra1][kk + 16 + half * 8];
            const int nb0 = wn * 32 + mrow;
            const int nb1 = nb0 + 16;
            b0.q[0] = *(const uint4*)&Btile[nb0][kk + half * 8];
            b0.q[1] = *(const uint4*)&Btile[nb0][kk + 16 + half * 8];
            b1.q[0] = *(const uint4*)&Btile[nb1][kk + half * 8];
            b1.q[1] = *(const uint4*)&Btile[nb1][kk + 16 + half * 8];

            acc00 = __builtin_amdgcn_wmma_f32_16x16x32_bf16(
                false, a0.v, false, b0.v, (short)0, acc00, false, false);
            acc01 = __builtin_amdgcn_wmma_f32_16x16x32_bf16(
                false, a0.v, false, b1.v, (short)0, acc01, false, false);
            acc10 = __builtin_amdgcn_wmma_f32_16x16x32_bf16(
                false, a1.v, false, b0.v, (short)0, acc10, false, false);
            acc11 = __builtin_amdgcn_wmma_f32_16x16x32_bf16(
                false, a1.v, false, b1.v, (short)0, acc11, false, false);
        }
        __syncthreads();
    }

    // C/D layout: VGPR r -> lanes 0-15: M=r, N=lane; lanes 16-31: M=8+r
#pragma unroll
    for (int r = 0; r < 8; ++r) {
        int r0 = wm * 32 + half * 8 + r;
        int r1 = r0 + 16;
        int c0 = ntile0 + wn * 32 + mrow;
        int c1 = c0 + 16;
        zout[(size_t)r0 * N_ + c0] = acc00[r];
        zout[(size_t)r0 * N_ + c1] = acc01[r];
        zout[(size_t)r1 * N_ + c0] = acc10[r];
        zout[(size_t)r1 * N_ + c1] = acc11[r];
    }
}

// ---------------------------------------------------------------------------
// Block reduction: shfl_xor within wave, 8-entry LDS across waves (2 barriers)
// ---------------------------------------------------------------------------
__device__ __forceinline__ float2 block_sum2(float2 v) {
    __shared__ float2 part[8];
#pragma unroll
    for (int m = 16; m > 0; m >>= 1) {
        v.x += __shfl_xor(v.x, m, 32);
        v.y += __shfl_xor(v.y, m, 32);
    }
    const int wave = threadIdx.x >> 5;
    const int lane = threadIdx.x & 31;
    if (lane == 0) part[wave] = v;
    __syncthreads();
    float2 r = part[0];
#pragma unroll
    for (int w = 1; w < 8; ++w) { r.x += part[w].x; r.y += part[w].y; }
    __syncthreads();
    return r;
}

// ---------------------------------------------------------------------------
// Pointwise: sum K-split halves, 4 gate layernorms + gate math + cell-state
// layernorm + masking; writes h (fp32 + bf16) and accumulates output mean.
// grid 64 (one block per batch row), block 256.
// ---------------------------------------------------------------------------
__global__ void __launch_bounds__(256) lstm_pointwise(
        const float* __restrict__ z,             // [2][B][N]
        float* __restrict__ c, float* __restrict__ h,
        unsigned short* __restrict__ hbf,
        const float* __restrict__ gamma, const float* __restrict__ beta,
        const int* __restrict__ lens, int t, int is_top,
        float* __restrict__ out) {
    const int b   = blockIdx.x;
    const int tid = threadIdx.x;
    const bool mask = t < lens[b];
    const float* z0 = z + (size_t)b * N_;
    const float* z1 = z + (size_t)B_ * N_ + (size_t)b * N_;
    const float inv_n = 1.0f / (float)H_;

    float g4[4][4];
#pragma unroll
    for (int g = 0; g < 4; ++g)
#pragma unroll
        for (int i = 0; i < 4; ++i) {
            int idx = g * H_ + tid + i * 256;
            g4[g][i] = z0[idx] + z1[idx];
        }

    // layernorm each gate independently over H
#pragma unroll
    for (int g = 0; g < 4; ++g) {
        float2 s = {0.0f, 0.0f};
#pragma unroll
        for (int i = 0; i < 4; ++i) {
            float v = g4[g][i];
            s.x += v;
            s.y += v * v;
        }
        float2 r = block_sum2(s);
        float mu  = r.x * inv_n;
        float var = r.y * inv_n - mu * mu;
        float inv = rsqrtf(var + 1e-12f);
#pragma unroll
        for (int i = 0; i < 4; ++i) {
            int hx = tid + i * 256;
            g4[g][i] = (g4[g][i] - mu) * inv * gamma[g * H_ + hx] + beta[g * H_ + hx];
        }
    }

    // gates -> raw new cell
    float co[4], ho[4], ncr[4];
    float2 s2 = {0.0f, 0.0f};
#pragma unroll
    for (int i = 0; i < 4; ++i) {
        int hx = tid + i * 256;
        co[i] = c[(size_t)b * H_ + hx];
        ho[i] = h[(size_t)b * H_ + hx];
        float iv = sigmoidf_(g4[0][i]);
        float jv = tanhf(g4[1][i]);
        float fv = sigmoidf_(g4[2][i] + 1.0f);   // FORGET_BIAS
        ncr[i] = co[i] * fv + iv * jv;
        s2.x += ncr[i];
        s2.y += ncr[i] * ncr[i];
    }
    float2 r2 = block_sum2(s2);
    float mu2  = r2.x * inv_n;
    float var2 = r2.y * inv_n - mu2 * mu2;
    float inv2 = rsqrtf(var2 + 1e-12f);

#pragma unroll
    for (int i = 0; i < 4; ++i) {
        int hx = tid + i * 256;
        float nc = (ncr[i] - mu2) * inv2 * gamma[4 * H_ + hx] + beta[4 * H_ + hx];
        float nh = tanhf(nc) * sigmoidf_(g4[3][i]);
        float cw = mask ? nc : co[i];
        float hw = mask ? nh : ho[i];
        c[(size_t)b * H_ + hx] = cw;
        h[(size_t)b * H_ + hx] = hw;
        hbf[(size_t)b * H_ + hx] = f2bf(hw);     // bf16 copy for next GEMM
        if (is_top) {
            out[(size_t)b * H_ + hx] += (mask ? nh : 0.0f) * (1.0f / (float)T_);
        }
    }
}

// ---------------------------------------------------------------------------
// Launch. Workspace: WbT 32MB | Xbf 32MB | hbf 256KB | c 512KB | h 512KB |
// z 2MB  (~67MB total). Graph-capture safe.
// ---------------------------------------------------------------------------
extern "C" void kernel_launch(void* const* d_in, const int* in_sizes, int n_in,
                              void* d_out, int out_size, void* d_ws, size_t ws_size,
                              hipStream_t stream) {
    (void)in_sizes; (void)n_in; (void)out_size; (void)ws_size;

    const float* inputs = (const float*)d_in[0];   // [B,T,H]
    const int*   lens   = (const int*)d_in[1];     // [B]
    const float* W      = (const float*)d_in[2];   // [L,2H,4H]
    const float* gamma  = (const float*)d_in[3];   // [L,5,H]
    const float* beta   = (const float*)d_in[4];   // [L,5,H]
    float*       out    = (float*)d_out;           // [B,H]

    char* ws = (char*)d_ws;
    unsigned short* WbT = (unsigned short*)ws;                   ws += (size_t)L_ * N_ * K_ * 2;
    unsigned short* Xbf = (unsigned short*)ws;                   ws += (size_t)B_ * T_ * H_ * 2;
    unsigned short* hbf = (unsigned short*)ws;                   ws += (size_t)L_ * B_ * H_ * 2;
    float* c = (float*)ws;                                       ws += (size_t)L_ * B_ * H_ * 4;
    float* h = (float*)ws;                                       ws += (size_t)L_ * B_ * H_ * 4;
    float* z = (float*)ws;                                       // [2][B][N]

    convert_x<<<(B_ * T_ * H_) / 1024, 256, 0, stream>>>(inputs, Xbf);
    prep_weights<<<dim3(N_ / 32, K_ / 32, L_), dim3(32, 8), 0, stream>>>(W, WbT);
    init_state<<<(L_ * B_ * H_) / 256, 256, 0, stream>>>(c, h, hbf, out);

    for (int t = 0; t < T_; ++t) {
        for (int l = 0; l < L_; ++l) {
            const unsigned short* A0 =
                (l == 0) ? (Xbf + (size_t)t * H_) : hbf;
            long sA0 = (l == 0) ? (long)T_ * H_ : (long)H_;
            const unsigned short* A1 = hbf + (size_t)l * B_ * H_;
            const unsigned short* Wl = WbT + (size_t)l * N_ * K_;

            lstm_gemm<<<dim3(N_ / 64, 2), 128, 0, stream>>>(A0, sA0, A1, Wl, z);
            lstm_pointwise<<<B_, 256, 0, stream>>>(
                z, c + (size_t)l * B_ * H_, h + (size_t)l * B_ * H_,
                hbf + (size_t)l * B_ * H_,
                gamma + (size_t)l * 5 * H_, beta + (size_t)l * 5 * H_,
                lens, t, (l == L_ - 1) ? 1 : 0, out);
        }
    }
}
